// GNNEncoder_4793183502594
// MI455X (gfx1250) — matrix operand order repaired and checked by reading, not compiled, not run on hardware
//
#include <hip/hip_runtime.h>
#include <hip/hip_bf16.h>

typedef __attribute__((ext_vector_type(16))) __bf16 v16bf;
typedef __attribute__((ext_vector_type(8)))  float  v8f;
typedef unsigned short u16t;

// ---------------- problem constants ----------------
constexpr int BGR   = 128;       // graphs
constexpr int NND   = 128;       // nodes per graph
constexpr int EMB   = 64;
constexpr int FFN   = 2048;
constexpr size_t NODES = (size_t)BGR * NND;          // 16384
constexpr size_t MT    = NODES * 8;                  // 131072 rows for LPE
constexpr size_t ADJ_TOT = 4ull * BGR * NND * NND;   // 8388608

// ---------------- LPE param blob offsets (jax pytree: dict keys sorted) ------
constexpr size_t LP_FF1B = 0,      LP_FF1W = 2048,   LP_FF2B = 133120, LP_FF2W = 133184;
constexpr size_t LP_INB  = 264256, LP_INW  = 264448, LP_LN1B = 276736, LP_LN1G = 276800;
constexpr size_t LP_LN2B = 276864, LP_LN2G = 276928, LP_OUTB = 276992, LP_OUTW = 277056;
constexpr size_t LP_STRIDE = 281152;
constexpr size_t LP_LINAB = 562304, LP_LINAW = 562368, LPE_TOT = 562496;

// ---------------- GNN param blob offsets (per bond: 3 layers of {b,bs,w,ws}) -
constexpr size_t GB_STRIDE = 82688, GNN_TOT = 330752;
constexpr size_t GB_B [3] = {0,     16640, 49664};
constexpr size_t GB_BS[3] = {128,   16768, 49792};
constexpr size_t GB_W [3] = {256,   16896, 49920};
constexpr size_t GB_WS[3] = {8448,  33280, 66304};
constexpr int    GDI [3] = {64, 128, 128};

// ---------------- workspace layout (bytes) ----------------
constexpr size_t alup(size_t x) { return (x + 255) & ~(size_t)255; }
constexpr size_t O_LPEBF = 0;
constexpr size_t O_GNNBF = alup(O_LPEBF + LPE_TOT * 2);
constexpr size_t O_ADJBF = alup(O_GNNBF + GNN_TOT * 2);
constexpr size_t O_VALID = alup(O_ADJBF + ADJ_TOT * 2);
constexpr size_t O_XAF   = alup(O_VALID + NODES * 4);
constexpr size_t O_XAH   = alup(O_XAF + MT * EMB * 4);
constexpr size_t O_XBF   = alup(O_XAH + MT * EMB * 2);
constexpr size_t O_XBH   = alup(O_XBF + MT * EMB * 4);
constexpr size_t O_QKV   = alup(O_XBH + MT * EMB * 2);
constexpr size_t O_OH    = alup(O_QKV + MT * 192 * 2);
constexpr size_t O_TF    = alup(O_OH + MT * EMB * 2);
constexpr size_t O_HBUF  = alup(O_TF + MT * EMB * 4);
constexpr size_t O_AE    = alup(O_HBUF + 16384ull * FFN * 2);
constexpr size_t O_AGG   = alup(O_AE + NODES * EMB * 2);
constexpr size_t O_LIN   = alup(O_AGG + NODES * 128 * 2);
constexpr size_t O_H0    = alup(O_LIN + NODES * 128 * 4);
constexpr size_t O_H1    = alup(O_H0 + NODES * 128 * 2);
constexpr size_t O_MOL   = alup(O_H1 + NODES * 128 * 2);

// ---------------- CDNA5 async global->LDS helpers ----------------
__device__ __forceinline__ unsigned lds_off(const void* p) {
  // generic LDS address = {SHARED_BASE, lds_byte_offset}; low 32 bits = DS address
  return (unsigned)(size_t)p;
}
__device__ __forceinline__ void async_ld16(unsigned dsaddr, const void* gaddr) {
  asm volatile("global_load_async_to_lds_b128 %0, %1, off"
               :: "v"(dsaddr), "v"(gaddr) : "memory");
}
__device__ __forceinline__ void wait_async0() {
  asm volatile("s_wait_asynccnt 0x0" ::: "memory");
}

__device__ __forceinline__ v16bf load_frag(const u16t* p) {
  union { uint4 q[2]; v16bf v; } u;
  u.q[0] = *(const uint4*)(p);
  u.q[1] = *(const uint4*)(p + 8);
  return u.v;
}

// =====================================================================
// Generic bf16 WMMA GEMM: C = act(A@B + bias + Cin), BM=128 BN=64 BK=32
// 256 threads = 8 waves; wave w owns rows [w*16, w*16+16), all 64 cols.
// Double-buffered LDS filled with global_load_async_to_lds_b128 (ASYNCcnt)
// so the DMA of tile i+1 overlaps the WMMA of tile i. Last iteration peeled
// so the steady-state loop body is branch-free.
// BSWZ=1: B pre-swizzled in global memory (fragment-ready 4KB tile blocks).
// =====================================================================
template <int BSWZ>
__global__ __launch_bounds__(256) void gemm_wmma(
    const __bf16* __restrict__ A, const __bf16* __restrict__ B,
    const float* __restrict__ bias, const float* __restrict__ Cin,
    float* __restrict__ Cf, __bf16* __restrict__ Ch,
    int M, int N, int K, long long sA, long long sB, long long sC, int relu)
{
  __shared__ __align__(16) u16t As[2][8 * 32 * 16];  // 2 x 8KB
  __shared__ __align__(16) u16t Bs[2][4 * 32 * 16];  // 2 x 4KB
  const int tid  = threadIdx.x;
  const int mw   = tid >> 5;
  const int lane = tid & 31;
  const int m0   = blockIdx.y * 128;
  const int n0   = blockIdx.x * 64;
  const __bf16* Ab = A + (size_t)blockIdx.z * sA;
  const __bf16* Bb = B + (size_t)blockIdx.z * sB;

  v8f acc[4];
#pragma unroll
  for (int t = 0; t < 4; ++t)
#pragma unroll
    for (int i = 0; i < 8; ++i) acc[t][i] = 0.0f;

  // ---- tile staging (async DMA into LDS, fragment-ready layout) ----
  auto stageA = [&](int buf, int ks) {
#pragma unroll
    for (int i = 0; i < 2; ++i) {
      int chunk = tid + i * 256;                 // 0..511
      int row   = chunk >> 2;                    // 0..127
      int kq    = (chunk & 3) * 8;               // 0,8,16,24
      const __bf16* gp = Ab + (size_t)(m0 + row) * K + (ks + kq);
      int lhi = (kq >= 16) ? 16 : 0;
      int h0  = ((kq == 8) || (kq == 24)) ? 8 : 0;
      async_ld16(lds_off(&As[buf][(((row >> 4) * 32) + lhi + (row & 15)) * 16 + h0]), gp);
    }
  };
  auto stageB = [&](int buf, int ks) {
    if (BSWZ) {  // pre-swizzled: straight 4KB block DMA
      const __bf16* gp = Bb + (((size_t)(ks >> 5) * (N >> 6)) + (n0 >> 6)) * 2048 + tid * 8;
      async_ld16(lds_off(&Bs[buf][tid * 8]), gp);
    } else {     // generic row-major B: load + transposing scatter into LDS
      int k  = tid >> 3;
      int nq = (tid & 7) * 8;
      uint4 d = *(const uint4*)(Bb + (size_t)(ks + k) * N + (n0 + nq));
      const u16t* hw = (const u16t*)&d;
      int lhi = (k & 16) ? 16 : 0;
      int kp  = k & 15;
#pragma unroll
      for (int j = 0; j < 8; ++j) {
        int n = nq + j;
        Bs[buf][(((n >> 4) * 32) + lhi + (n & 15)) * 16 + kp] = hw[j];
      }
    }
  };
  // preload A + all 4 B frags, then chain 4 WMMAs (partial dscnt waits pipeline)
  auto compute = [&](int cur) {
    v16bf af = load_frag(&As[cur][(mw * 32 + lane) * 16]);
    v16bf bfr[4];
#pragma unroll
    for (int t = 0; t < 4; ++t)
      bfr[t] = load_frag(&Bs[cur][(t * 32 + lane) * 16]);
#pragma unroll
    for (int t = 0; t < 4; ++t)
      acc[t] = __builtin_amdgcn_wmma_f32_16x16x32_bf16(
          false, af, false, bfr[t], (short)0, acc[t], false, false);
  };

  const int nk = K >> 5;
  stageA(0, 0);
  stageB(0, 0);
  wait_async0();
  __syncthreads();

  for (int i = 0; i < nk - 1; ++i) {   // steady state: branch-free body
    const int cur = i & 1;
    stageA(cur ^ 1, (i + 1) << 5);     // DMA next tile, overlaps WMMA below
    stageB(cur ^ 1, (i + 1) << 5);
    compute(cur);
    wait_async0();                     // next tile landed
    __syncthreads();                   // publish (also flushes DS reads)
  }
  compute((nk - 1) & 1);               // peeled last tile

  // ---- epilogue: C tile layout lane<16 -> M=r, lane>=16 -> M=r+8, N=lane&15
  const float* Cib = Cin ? Cin + (size_t)blockIdx.z * sC : nullptr;
  float*  Cfb = Cf ? Cf + (size_t)blockIdx.z * sC : nullptr;
  __bf16* Chb = Ch ? Ch + (size_t)blockIdx.z * sC : nullptr;
  const int rbase = m0 + mw * 16 + ((lane & 16) ? 8 : 0);
  const int cl = lane & 15;
#pragma unroll
  for (int t = 0; t < 4; ++t) {
    int col = n0 + t * 16 + cl;
    float bv = bias ? bias[col] : 0.0f;
#pragma unroll
    for (int r = 0; r < 8; ++r) {
      size_t off = (size_t)(rbase + r) * N + col;
      float v = acc[t][r] + bv;
      if (Cib) v += Cib[off];
      if (relu) v = v > 0.0f ? v : 0.0f;
      if (Cfb) Cfb[off] = v;
      if (Chb) Chb[off] = (__bf16)v;
    }
  }
}

// ---------------- weight pre-swizzle: row-major f32 [K,N] -> fragment-ready
// bf16 blocks of 2048 elems per (32k x 64n) tile, tiles row-major over (kt,nt)
__global__ void swz_b_kernel(const float* __restrict__ src, __bf16* __restrict__ dst,
                             int K, int N) {
  int idx = blockIdx.x * blockDim.x + threadIdx.x;
  if (idx >= K * N) return;
  int k = idx / N, n = idx % N;
  int kl = k & 31, nl = n & 63;
  size_t blk = ((size_t)(k >> 5) * (N >> 6) + (n >> 6)) * 2048;
  size_t pos = (size_t)(((nl >> 4) * 32) + ((kl & 16) ? 16 : 0) + (nl & 15)) * 16 + (kl & 15);
  dst[blk + pos] = (__bf16)src[idx];
}

// ---------------- pointwise / small kernels ----------------
__global__ void cvt_bf16_kernel(const float* __restrict__ s, __bf16* __restrict__ d, int n) {
  int i = blockIdx.x * blockDim.x + threadIdx.x;
  if (i < n) d[i] = (__bf16)s[i];
}

__global__ void embed_kernel(const float* __restrict__ vec, const float* __restrict__ val,
                             const float* __restrict__ w, const float* __restrict__ b,
                             float* __restrict__ xf, __bf16* __restrict__ xh,
                             int* __restrict__ valid) {
  int idx = blockIdx.x * blockDim.x + threadIdx.x;   // MT * 64
  int e = idx & 63;
  int rf = idx >> 6;                                 // node*8 + f
  float a = vec[rf], c = val[rf];
  bool nan = (a != a);
  if ((idx & 511) == 0) valid[rf >> 3] = nan ? 0 : 1;
  if (nan) a = 0.0f;
  if (c != c) c = 0.0f;
  float y = a * w[e] + c * w[64 + e] + b[e];
  xf[idx] = y;
  xh[idx] = (__bf16)y;
}

__global__ void attn_kernel(const __bf16* __restrict__ qkv, const int* __restrict__ valid,
                            __bf16* __restrict__ oh) {
  int idx = blockIdx.x * blockDim.x + threadIdx.x;   // NODES*32
  int node = idx >> 5;
  int head = (idx >> 3) & 3;
  int f1   = idx & 7;
  const __bf16* base = qkv + (size_t)node * 8 * 192;
  int ho = head * 16;
  float q[16];
#pragma unroll
  for (int d = 0; d < 16; ++d) q[d] = (float)base[f1 * 192 + ho + d];
  bool inval = (valid[node] == 0);
  float s[8], mx = -3.4e38f;
#pragma unroll
  for (int f2 = 0; f2 < 8; ++f2) {
    float a = 0.0f;
#pragma unroll
    for (int d = 0; d < 16; ++d) a += q[d] * (float)base[f2 * 192 + 64 + ho + d];
    s[f2] = inval ? -1e9f : a * 0.25f;
    mx = fmaxf(mx, s[f2]);
  }
  float den = 0.0f;
#pragma unroll
  for (int f2 = 0; f2 < 8; ++f2) { s[f2] = __expf(s[f2] - mx); den += s[f2]; }
  float inv = 1.0f / den;
  float o[16];
#pragma unroll
  for (int d = 0; d < 16; ++d) o[d] = 0.0f;
#pragma unroll
  for (int f2 = 0; f2 < 8; ++f2) {
    float a = s[f2] * inv;
#pragma unroll
    for (int d = 0; d < 16; ++d) o[d] += a * (float)base[f2 * 192 + 128 + ho + d];
  }
  __bf16* op = oh + ((size_t)node * 8 + f1) * 64 + ho;
#pragma unroll
  for (int d = 0; d < 16; ++d) op[d] = (__bf16)o[d];
}

__global__ __launch_bounds__(256) void add_ln_kernel(
    const float* __restrict__ resid, const float* __restrict__ add,
    const float* __restrict__ g, const float* __restrict__ b,
    float* __restrict__ of, __bf16* __restrict__ oh) {
  int wave = threadIdx.x >> 5, lane = threadIdx.x & 31;
  size_t row = (size_t)blockIdx.x * 8 + wave;
  const float* r0 = resid + row * 64;
  const float* a0 = add + row * 64;
  float x1 = r0[lane] + a0[lane];
  float x2 = r0[lane + 32] + a0[lane + 32];
  float sum = x1 + x2, sq = x1 * x1 + x2 * x2;
#pragma unroll
  for (int m = 16; m >= 1; m >>= 1) {
    sum += __shfl_xor(sum, m, 32);
    sq  += __shfl_xor(sq, m, 32);
  }
  float mean = sum * (1.0f / 64.0f);
  float var  = sq * (1.0f / 64.0f) - mean * mean;
  float rs = rsqrtf(var + 1e-5f);
  float y1 = (x1 - mean) * rs * g[lane] + b[lane];
  float y2 = (x2 - mean) * rs * g[lane + 32] + b[lane + 32];
  of[row * 64 + lane] = y1;         of[row * 64 + lane + 32] = y2;
  oh[row * 64 + lane] = (__bf16)y1; oh[row * 64 + lane + 32] = (__bf16)y2;
}

__global__ void ae_kernel(const float* __restrict__ xf, const int* __restrict__ valid,
                          __bf16* __restrict__ aeh) {
  int idx = blockIdx.x * blockDim.x + threadIdx.x;   // NODES*64
  int node = idx >> 6, e = idx & 63;
  float s = 0.0f;
  if (valid[node]) {
#pragma unroll
    for (int f = 0; f < 8; ++f) s += xf[((size_t)node * 8 + f) * 64 + e];
  }
  aeh[idx] = (__bf16)s;
}

__global__ void fill_f32_kernel(float* __restrict__ p, size_t n) {
  size_t i = (size_t)blockIdx.x * blockDim.x + threadIdx.x;
  if (i < n) p[i] = 0.0f;
}

__global__ void selu_acc_kernel(const float* __restrict__ h, float* __restrict__ mol, int off) {
  int idx = blockIdx.x * blockDim.x + threadIdx.x;   // NODES*128
  int row = idx >> 7, c = idx & 127;
  float v = h[idx];
  const float sc = 1.0507009873554805f, al = 1.6732632423543772f;
  float y = v > 0.0f ? sc * v : sc * al * (__expf(v) - 1.0f);
  mol[(size_t)row * 384 + off + c] += y;
}

__global__ void glo_kernel(const float* __restrict__ mol, const int* __restrict__ valid,
                           float* __restrict__ out) {
  int idx = blockIdx.x * blockDim.x + threadIdx.x;   // B*384
  int b = idx / 384, c = idx % 384;
  float s = 0.0f;
  for (int n = 0; n < NND; ++n)
    if (valid[b * NND + n]) s += mol[((size_t)b * NND + n) * 384 + c];
  out[idx] = s;
}

__global__ void loc_kernel(const float* __restrict__ mol, const int* __restrict__ valid,
                           float* __restrict__ out) {
  size_t idx = (size_t)blockIdx.x * blockDim.x + threadIdx.x;  // NODES*384
  size_t row = idx / 384;
  out[idx] = valid[row] ? mol[idx] : __uint_as_float(0x7fc00000u);
}

// ---------------- host orchestration ----------------
static void launch_gemm(hipStream_t s, const void* A, const void* B, const float* bias,
                        const float* Cin, float* Cf, void* Ch,
                        int M, int N, int K, int batch,
                        long long sA, long long sB, long long sC, int relu, int bswz) {
  dim3 g((unsigned)(N / 64), (unsigned)(M / 128), (unsigned)batch);
  if (bswz)
    gemm_wmma<1><<<g, 256, 0, s>>>((const __bf16*)A, (const __bf16*)B, bias, Cin, Cf,
                                   (__bf16*)Ch, M, N, K, sA, sB, sC, relu);
  else
    gemm_wmma<0><<<g, 256, 0, s>>>((const __bf16*)A, (const __bf16*)B, bias, Cin, Cf,
                                   (__bf16*)Ch, M, N, K, sA, sB, sC, relu);
}
static void launch_swz(hipStream_t s, const float* src, void* dst, int K, int N) {
  int n = K * N;
  swz_b_kernel<<<(n + 255) / 256, 256, 0, s>>>(src, (__bf16*)dst, K, N);
}

extern "C" void kernel_launch(void* const* d_in, const int* in_sizes, int n_in,
                              void* d_out, int out_size, void* d_ws, size_t ws_size,
                              hipStream_t stream) {
  (void)in_sizes; (void)n_in; (void)out_size; (void)ws_size;
  const float* adjsF  = (const float*)d_in[1];
  const float* eigVec = (const float*)d_in[2];
  const float* eigVal = (const float*)d_in[3];
  const float* lpeF   = (const float*)d_in[5];
  const float* gnnF   = (const float*)d_in[6];
  float* out = (float*)d_out;
  char* ws = (char*)d_ws;

  __bf16* lpeH  = (__bf16*)(ws + O_LPEBF);   // swizzled weight images (same offsets)
  __bf16* gnnH  = (__bf16*)(ws + O_GNNBF);
  __bf16* adjH  = (__bf16*)(ws + O_ADJBF);
  int*    valid = (int*)(ws + O_VALID);
  float*  xaf = (float*)(ws + O_XAF);   __bf16* xah = (__bf16*)(ws + O_XAH);
  float*  xbf = (float*)(ws + O_XBF);   __bf16* xbh = (__bf16*)(ws + O_XBH);
  __bf16* qkvh = (__bf16*)(ws + O_QKV);
  __bf16* oh   = (__bf16*)(ws + O_OH);
  float*  tf   = (float*)(ws + O_TF);
  __bf16* Hbuf = (__bf16*)(ws + O_HBUF);
  __bf16* aeh  = (__bf16*)(ws + O_AE);
  __bf16* aggh = (__bf16*)(ws + O_AGG);
  float*  linf = (float*)(ws + O_LIN);
  __bf16* h0h  = (__bf16*)(ws + O_H0);
  __bf16* h1h  = (__bf16*)(ws + O_H1);
  float*  mol  = (float*)(ws + O_MOL);

  // ---- pre-swizzle all weight matrices into fragment-ready bf16 layout ----
  for (int L = 0; L < 2; ++L) {
    size_t lb = (size_t)L * LP_STRIDE;
    launch_swz(stream, lpeF + lb + LP_INW,  lpeH + lb + LP_INW,  64, 192);
    launch_swz(stream, lpeF + lb + LP_OUTW, lpeH + lb + LP_OUTW, 64, 64);
    launch_swz(stream, lpeF + lb + LP_FF1W, lpeH + lb + LP_FF1W, 64, FFN);
    launch_swz(stream, lpeF + lb + LP_FF2W, lpeH + lb + LP_FF2W, FFN, 64);
  }
  for (int bond = 0; bond < 4; ++bond) {
    size_t G = (size_t)bond * GB_STRIDE;
    for (int li = 0; li < 3; ++li) {
      launch_swz(stream, gnnF + G + GB_W[li],  gnnH + G + GB_W[li],  GDI[li], 128);
      launch_swz(stream, gnnF + G + GB_WS[li], gnnH + G + GB_WS[li], GDI[li], 128);
    }
  }
  // adjacency to bf16 (used as A matrices, plain row-major)
  cvt_bf16_kernel<<<((int)ADJ_TOT + 255) / 256, 256, 0, stream>>>(adjsF, adjH, (int)ADJ_TOT);

  // ---- LearnedPE input projection ----
  embed_kernel<<<(unsigned)(MT * 64 / 256), 256, 0, stream>>>(
      eigVec, eigVal, lpeF + LP_LINAW, lpeF + LP_LINAB, xaf, xah, valid);

  // ---- 2 transformer encoder layers ----
  for (int L = 0; L < 2; ++L) {
    size_t lb = (size_t)L * LP_STRIDE;
    launch_gemm(stream, xah, lpeH + lb + LP_INW, lpeF + lb + LP_INB, nullptr,
                nullptr, qkvh, (int)MT, 192, 64, 1, 0, 0, 0, 0, 1);
    attn_kernel<<<(unsigned)(NODES * 32 / 256), 256, 0, stream>>>(qkvh, valid, oh);
    launch_gemm(stream, oh, lpeH + lb + LP_OUTW, lpeF + lb + LP_OUTB, nullptr,
                tf, nullptr, (int)MT, 64, 64, 1, 0, 0, 0, 0, 1);
    add_ln_kernel<<<(unsigned)(MT / 8), 256, 0, stream>>>(
        xaf, tf, lpeF + lb + LP_LN1G, lpeF + lb + LP_LN1B, xbf, xbh);
    // FFN chunked so the 64MB hidden tile stays resident in the 192MB L2
    for (int ch = 0; ch < 8; ++ch) {
      size_t ro = (size_t)ch * 16384 * 64;
      launch_gemm(stream, xbh + ro, lpeH + lb + LP_FF1W, lpeF + lb + LP_FF1B, nullptr,
                  nullptr, Hbuf, 16384, FFN, 64, 1, 0, 0, 0, 1, 1);
      launch_gemm(stream, Hbuf, lpeH + lb + LP_FF2W, lpeF + lb + LP_FF2B, nullptr,
                  tf + ro, nullptr, 16384, 64, FFN, 1, 0, 0, 0, 0, 1);
    }
    add_ln_kernel<<<(unsigned)(MT / 8), 256, 0, stream>>>(
        xbf, tf, lpeF + lb + LP_LN2G, lpeF + lb + LP_LN2B, xaf, xah);
  }

  // ---- atom embeddings: nansum over freq ----
  ae_kernel<<<(unsigned)(NODES * 64 / 256), 256, 0, stream>>>(xaf, valid, aeh);

  // ---- GNN: 4 bonds x 3 GraphConvSkip layers ----
  fill_f32_kernel<<<(unsigned)(NODES * 384 / 256), 256, 0, stream>>>(mol, NODES * 384);
  for (int bond = 0; bond < 4; ++bond) {
    const __bf16* adjb = adjH + (size_t)bond * BGR * NND * NND;
    size_t G = (size_t)bond * GB_STRIDE;
    const __bf16* hin = aeh;
    __bf16* ping = h0h; __bf16* pong = h1h;
    for (int li = 0; li < 3; ++li) {
      int di = GDI[li];
      // agg = adj @ h  (batched over 128 graphs; B = activations -> generic path)
      launch_gemm(stream, adjb, hin, nullptr, nullptr, nullptr, aggh,
                  NND, di, NND, BGR,
                  (long long)NND * NND, (long long)NND * di, (long long)NND * di, 0, 0);
      // lin = agg @ W + b
      launch_gemm(stream, aggh, gnnH + G + GB_W[li], gnnF + G + GB_B[li], nullptr,
                  linf, nullptr, (int)NODES, 128, di, 1, 0, 0, 0, 0, 1);
      // h' = lin + h @ Ws + bs
      launch_gemm(stream, hin, gnnH + G + GB_WS[li], gnnF + G + GB_BS[li], linf,
                  linf, ping, (int)NODES, 128, di, 1, 0, 0, 0, 0, 1);
      selu_acc_kernel<<<(unsigned)(NODES * 128 / 256), 256, 0, stream>>>(linf, mol, li * 128);
      hin = ping;
      __bf16* tmp = ping; ping = pong; pong = tmp;
    }
  }

  // ---- outputs: glo [B,384] then loc [B*N,384] ----
  glo_kernel<<<(unsigned)(BGR * 384 / 256), 256, 0, stream>>>(mol, valid, out);
  loc_kernel<<<(unsigned)(NODES * 384 / 256), 256, 0, stream>>>(mol, valid, out + BGR * 384);
}
